// LSTMAttention_11639361372654
// MI455X (gfx1250) — matrix operand order repaired and checked by reading, not compiled, not run on hardware
//
#include <hip/hip_runtime.h>
#include <hip/hip_bf16.h>
#include <math.h>

typedef __attribute__((ext_vector_type(16))) __bf16 v16bf;
typedef __attribute__((ext_vector_type(8)))  float  v8f;

#define B_    256
#define L_    196
#define D_    2048
#define A_    512
#define LTILE 32               // L rows per batch element per WG
#define NB    2                // batch elements per WG
#define ROWS  (LTILE * NB)     // 64 LDS rows
#define LDSS  2056             // LDS row stride in bf16 elements (2048 + 8 pad -> 4-bank rotation/row)

static __device__ __forceinline__ unsigned short f2bf(float f) {
  unsigned int u = __float_as_uint(f);
  u += 0x7FFFu + ((u >> 16) & 1u);           // round-to-nearest-even
  return (unsigned short)(u >> 16);
}

union BFrag { uint4 q[2]; v16bf v; };

// ---------------- Kernel 0: W_w fp32 [512x2048] -> bf16 (lives in L2) ----------------
__global__ void k_convert_w(const float* __restrict__ Ww, unsigned int* __restrict__ Wbf) {
  int idx = blockIdx.x * blockDim.x + threadIdx.x;          // bf16-pair index
  float2 v = ((const float2*)Ww)[idx];
  Wbf[idx] = (unsigned int)f2bf(v.x) | ((unsigned int)f2bf(v.y) << 16);
}

// ---------------- Kernel 1: U_h[b,a] = hidden[b,:] . U_w[a,:] + U_b[a] ----------------
__global__ void k_uh(const float* __restrict__ hid, const float* __restrict__ Uw,
                     const float* __restrict__ Ub, float* __restrict__ Uh) {
  __shared__ float sh[A_];
  int b = blockIdx.x, t = threadIdx.x;
  sh[t] = hid[b * A_ + t];
  __syncthreads();
  const float* up = Uw + (size_t)t * A_;
  float acc = Ub[t];
#pragma unroll 8
  for (int k = 0; k < A_; ++k) acc = fmaf(up[k], sh[k], acc);
  Uh[b * A_ + t] = acc;
}

// ---------------- Kernel 2: fused W_s GEMM + tanh + v-dot -> e[b,l] ----------------
// Grid: (ceil(L/32), B/2). Block: 512 threads = 16 waves.
// WG covers TWO batch elements' 32-row L-tiles (64 LDS rows) so the 2 MB bf16 W_w
// is pulled from L2 once per 64 rows instead of once per 32.
// Wave w: m-tile = w&3 (rows mtile*16..+15; mtile 0,1 -> b0, 2,3 -> b1),
//         n-group = w>>2 (128 attention dims = 8 WMMA n-tiles).
__global__ __launch_bounds__(512) void k_att_scores(
    const float* __restrict__ img, const unsigned short* __restrict__ Wbf,
    const float* __restrict__ Wb,  const float* __restrict__ Uh,
    const float* __restrict__ vw,  const float* __restrict__ vb,
    float* __restrict__ eout) {
  __shared__ unsigned short s_img[ROWS * LDSS];             // 263,168 B
  __shared__ float s_e[ROWS];

  const int tid    = threadIdx.x;
  const int b_base = blockIdx.y * NB;
  const int l0     = blockIdx.x * LTILE;

  if (tid < ROWS) s_e[tid] = 0.0f;

  // Stage img tiles (2 x 32 x 2048) as bf16 into LDS; rows beyond L zero-padded.
  for (int idx = tid; idx < ROWS * (D_ / 2); idx += 512) {
    int row = idx >> 10;                                    // D_/2 = 1024 pairs per row
    int kp  = idx & 1023;
    int b   = b_base + (row >> 5);
    int l   = l0 + (row & 31);
    float2 v = make_float2(0.0f, 0.0f);
    if (l < L_) v = *(const float2*)(img + ((size_t)b * L_ + l) * D_ + 2 * kp);
    unsigned int p = (unsigned int)f2bf(v.x) | ((unsigned int)f2bf(v.y) << 16);
    *(unsigned int*)(&s_img[row * LDSS + 2 * kp]) = p;
  }
  __syncthreads();

  const int lane  = tid & 31;
  const int wave  = tid >> 5;
  const int mtile = wave & 3;        // 0..3 -> 16-row blocks; b = b_base + (mtile>>1)
  const int ngrp  = wave >> 2;       // 0..3 -> 128 attention dims
  const int half  = lane >> 4;       // K-half selector
  const int l16   = lane & 15;

  v8f acc[8] = {};

  // A fragment: row = mtile*16 + l16; elements {0..7} = K+{0..7|8..15}, {8..15} = K+{16..23|24..31}
  const unsigned short* ap  = &s_img[(mtile * 16 + l16) * LDSS + half * 8];
  // B fragment: column a = ngrp*128 + n*16 + l16; elements {0..7} = K+{0..7|8..15}, {8..15} = K+{16..23|24..31}
  const unsigned short* bp0 = Wbf + (size_t)(ngrp * 128 + l16) * D_ + half * 8;

  for (int k0 = 0; k0 < D_; k0 += 32) {
    BFrag af;
    af.q[0] = *(const uint4*)(ap + k0);
    af.q[1] = *(const uint4*)(ap + k0 + 16);
#pragma unroll
    for (int n = 0; n < 8; ++n) {
      BFrag bf;
      const unsigned short* bp = bp0 + (size_t)(n * 16) * D_ + k0;
      bf.q[0] = *(const uint4*)(bp);
      bf.q[1] = *(const uint4*)(bp + 16);
      acc[n] = __builtin_amdgcn_wmma_f32_16x16x32_bf16(
          /*neg_a=*/false, af.v, /*neg_b=*/false, bf.v,
          /*c_mod=*/(short)0, acc[n], /*reuse_a=*/false, /*reuse_b=*/false);
    }
  }

  // Epilogue: e_partial[row] += v[a] * tanh(W_s + W_b[a] + U_h[b,a])
  const int bq = b_base + (mtile >> 1);
  float ep[8];
#pragma unroll
  for (int i = 0; i < 8; ++i) ep[i] = 0.0f;
#pragma unroll
  for (int n = 0; n < 8; ++n) {
    int a = ngrp * 128 + n * 16 + l16;
    float bias = Wb[a] + Uh[(size_t)bq * A_ + a];
    float vv   = vw[a];
#pragma unroll
    for (int i = 0; i < 8; ++i) {
      float t = tanhf(acc[n][i] + bias);
      ep[i] = fmaf(t, vv, ep[i]);
    }
  }
  // Sum across the 16 lanes of each half (C layout: VGPR i -> row i (lanes 0-15) / i+8 (lanes 16-31))
#pragma unroll
  for (int m = 8; m >= 1; m >>= 1) {
#pragma unroll
    for (int i = 0; i < 8; ++i) ep[i] += __shfl_xor(ep[i], m, 16);
  }
  if (l16 == 0) {
    int rbase = mtile * 16 + half * 8;
#pragma unroll
    for (int i = 0; i < 8; ++i) atomicAdd(&s_e[rbase + i], ep[i]);
  }
  __syncthreads();

  if (tid < ROWS) {
    int b = b_base + (tid >> 5);
    int l = l0 + (tid & 31);
    if (l < L_) eout[(size_t)b * L_ + l] = s_e[tid] + vb[0];
  }
}

// ---------------- Kernel 3: softmax over L per batch ----------------
__global__ void k_softmax(const float* __restrict__ e, float* __restrict__ alpha) {
  __shared__ float red[256];
  int b = blockIdx.x, t = threadIdx.x;
  float x = (t < L_) ? e[(size_t)b * L_ + t] : -1e30f;
  red[t] = x;
  __syncthreads();
  for (int s = 128; s > 0; s >>= 1) { if (t < s) red[t] = fmaxf(red[t], red[t + s]); __syncthreads(); }
  float mx = red[0];
  __syncthreads();
  float ex = (t < L_) ? __expf(x - mx) : 0.0f;
  red[t] = ex;
  __syncthreads();
  for (int s = 128; s > 0; s >>= 1) { if (t < s) red[t] += red[t + s]; __syncthreads(); }
  float inv = 1.0f / red[0];
  if (t < L_) alpha[(size_t)b * L_ + t] = ex * inv;
}

// ---------------- Kernel 4: context[b,d] = sum_l img[b,l,d] * alpha[b,l] ----------------
__global__ void k_context(const float* __restrict__ img, const float* __restrict__ alpha,
                          float* __restrict__ ctx) {
  __shared__ float sa[L_];
  int b = blockIdx.y, t = threadIdx.x;
  if (t < L_) sa[t] = alpha[(size_t)b * L_ + t];
  __syncthreads();
  int d = blockIdx.x * 256 + t;
  const float* ip = img + (size_t)b * L_ * D_ + d;
  float acc = 0.0f;
#pragma unroll 4
  for (int l = 0; l < L_; ++l) acc = fmaf(ip[(size_t)l * D_], sa[l], acc);
  ctx[(size_t)b * D_ + d] = acc;
}

// ---------------- Launch ----------------
extern "C" void kernel_launch(void* const* d_in, const int* in_sizes, int n_in,
                              void* d_out, int out_size, void* d_ws, size_t ws_size,
                              hipStream_t stream) {
  const float* img = (const float*)d_in[0];
  const float* hid = (const float*)d_in[1];
  const float* Uw  = (const float*)d_in[2];
  const float* Ub  = (const float*)d_in[3];
  const float* Ww  = (const float*)d_in[4];
  const float* Wb  = (const float*)d_in[5];
  const float* vw  = (const float*)d_in[6];
  const float* vb  = (const float*)d_in[7];

  // workspace layout: [W_w bf16: 2 MB][U_h: 512 KB][e: 200 KB]  (~2.8 MB total)
  char* ws = (char*)d_ws;
  unsigned int*   Wbf32 = (unsigned int*)ws;
  unsigned short* Wbf   = (unsigned short*)ws;
  float* Uh = (float*)(ws + (size_t)A_ * D_ * 2);
  float* eW = (float*)(ws + (size_t)A_ * D_ * 2 + (size_t)B_ * A_ * 4);

  float* ctx   = (float*)d_out;                 // [256, 2048]
  float* alpha = ctx + (size_t)B_ * D_;         // [256, 196]

  k_convert_w <<<(A_ * D_ / 2) / 256, 256, 0, stream>>>(Ww, Wbf32);
  k_uh        <<<B_, A_, 0, stream>>>(hid, Uw, Ub, Uh);
  k_att_scores<<<dim3((L_ + LTILE - 1) / LTILE, B_ / NB), 512, 0, stream>>>(img, Wbf, Wb, Uh, vw, vb, eW);
  k_softmax   <<<B_, 256, 0, stream>>>(eW, alpha);
  k_context   <<<dim3(D_ / 256, B_), 256, 0, stream>>>(img, alpha, ctx);
}
